// GBC_10273561772571
// MI455X (gfx1250) — compile-verified, matrix operand order checked by reading, not run on hardware
//
#include <hip/hip_runtime.h>
#include <hip/hip_bf16.h>

// ---------------------------------------------------------------------------
// CondConv dynamic convolution for MI455X (gfx1250, wave32, WMMA + TDM).
// Pipeline: pool -> attention -> f16 padded-x / tap-major f16 agg weights ->
// implicit-GEMM 3x3 conv with v_wmma_f32_16x16x32_f16 (f32 accum),
// A-tiles staged via the Tensor Data Mover (tensor_load_to_lds, FORCED ON
// this round to probe the builtin), with s_wait_tensorcnt synchronization.
// ---------------------------------------------------------------------------

#define NB    16      // batch
#define CREF  256
#define CIN   256
#define COUT  256
#define KK    4
#define HID   65
#define HH    56
#define WW    56
#define PP    (HH*WW)      // 3136 spatial positions
#define HP    58
#define SP    (HP*HP)      // 3364 padded positions
#define NTAP  9
#define TEMP  34.0f

typedef __attribute__((ext_vector_type(16))) _Float16 v16h;
typedef __attribute__((ext_vector_type(8)))  _Float16 v8h;
typedef __attribute__((ext_vector_type(8)))  float    v8f;
typedef __attribute__((ext_vector_type(4)))  unsigned int v4u;
typedef __attribute__((ext_vector_type(8)))  int      v8i;
typedef __attribute__((ext_vector_type(4)))  int      v4i;

// ---------------- workspace layout (bytes) ----------------
#define WS_POOLED 0                                   // 16*256 f32
#define WS_ATTN   16384                               // 16*4  f32
#define WS_AGGB   32768                               // 16*256 f32
#define WS_XH     65536                               // 16*256*3364 f16 (padded)
#define WS_WH     (WS_XH + (size_t)NB*CIN*SP*2)       // 16*9*256*256 f16 (tap-major)
// total ~44.4 MB

// ---------------- 1) adaptive avg-pool over HxW ----------------
__global__ __launch_bounds__(256) void pool_kernel(const float* __restrict__ ref,
                                                   float* __restrict__ pooled) {
  const int bc = blockIdx.x;                 // b*CREF + c
  const float* src = ref + (size_t)bc * PP;
  float s = 0.f;
  for (int i = threadIdx.x; i < PP; i += 256) s += src[i];
  __shared__ float red[256];
  red[threadIdx.x] = s;
  __syncthreads();
  for (int st = 128; st > 0; st >>= 1) {
    if (threadIdx.x < st) red[threadIdx.x] += red[threadIdx.x + st];
    __syncthreads();
  }
  if (threadIdx.x == 0) pooled[bc] = red[0] * (1.f / (float)PP);
}

// ---------------- 2) attention + aggregated bias (single block) ----------------
__global__ __launch_bounds__(256) void attn_kernel(const float* __restrict__ pooled,
                                                   const float* __restrict__ fc1_w,
                                                   const float* __restrict__ fc2_w,
                                                   const float* __restrict__ fc2_b,
                                                   const float* __restrict__ bias,
                                                   float* __restrict__ attn,
                                                   float* __restrict__ aggB) {
  __shared__ float sh[NB * HID];     // relu(fc1)
  __shared__ float slog[NB * KK];
  __shared__ float sattn[NB * KK];
  const int tid = threadIdx.x;

  for (int idx = tid; idx < NB * HID; idx += 256) {
    const int b = idx / HID, j = idx % HID;
    float acc = 0.f;
    const float* p = pooled + b * CREF;
    const float* w = fc1_w + j * CREF;
    for (int c = 0; c < CREF; ++c) acc += p[c] * w[c];
    sh[idx] = acc > 0.f ? acc : 0.f;
  }
  __syncthreads();

  for (int idx = tid; idx < NB * KK; idx += 256) {
    const int b = idx / KK, k = idx % KK;
    float acc = fc2_b[k];
    const float* h = sh + b * HID;
    const float* w = fc2_w + k * HID;
    for (int j = 0; j < HID; ++j) acc += h[j] * w[j];
    slog[idx] = acc * (1.0f / TEMP);
  }
  __syncthreads();

  if (tid < NB) {
    const int b = tid;
    float m = slog[b * KK];
    for (int k = 1; k < KK; ++k) m = fmaxf(m, slog[b * KK + k]);
    float s = 0.f, e[KK];
    for (int k = 0; k < KK; ++k) { e[k] = __expf(slog[b * KK + k] - m); s += e[k]; }
    const float inv = 1.f / s;
    for (int k = 0; k < KK; ++k) {
      sattn[b * KK + k] = e[k] * inv;
      attn[b * KK + k]  = e[k] * inv;
    }
  }
  __syncthreads();

  for (int idx = tid; idx < NB * COUT; idx += 256) {
    const int b = idx / COUT, co = idx % COUT;
    float acc = 0.f;
    for (int k = 0; k < KK; ++k) acc += sattn[b * KK + k] * bias[k * COUT + co];
    aggB[idx] = acc;
  }
}

// ---------------- 3) x -> zero-padded f16 image [b][c][58][58] ----------------
__global__ __launch_bounds__(256) void cvt_x_kernel(const float* __restrict__ x,
                                                    _Float16* __restrict__ Xh) {
  const size_t idx = (size_t)blockIdx.x * 256 + threadIdx.x;
  const size_t total = (size_t)NB * CIN * SP;
  if (idx >= total) return;
  const int sp = (int)(idx % SP);
  const size_t bc = idx / SP;
  const int yy = sp / HP, xx = sp % HP;
  float v = 0.f;
  if (yy >= 1 && yy <= HH && xx >= 1 && xx <= WW)
    v = x[bc * PP + (size_t)(yy - 1) * WW + (xx - 1)];
  Xh[idx] = (_Float16)v;
}

// ---------------- 4) aggregate weights, tap-major f16: Wh[b][tap][co][ci] ----
__global__ __launch_bounds__(256) void agg_w_kernel(const float* __restrict__ weight,
                                                    const float* __restrict__ attn,
                                                    _Float16* __restrict__ Wh) {
  const size_t idx = (size_t)blockIdx.x * 256 + threadIdx.x;
  const size_t total = (size_t)NB * NTAP * COUT * CIN;
  if (idx >= total) return;
  const int ci  = (int)(idx % CIN);
  size_t r = idx / CIN;
  const int co  = (int)(r % COUT); r /= COUT;
  const int tap = (int)(r % NTAP);
  const int b   = (int)(r / NTAP);
  float acc = 0.f;
  for (int k = 0; k < KK; ++k) {
    const size_t wi = (((size_t)k * COUT + co) * CIN + ci) * NTAP + tap;
    acc += attn[b * KK + k] * weight[wi];
  }
  Wh[idx] = (_Float16)acc;
}

// ---------------- 5) implicit-GEMM conv with WMMA + TDM A staging ----------
// Per batch: D[COUT x 3136] = Wh[b] (256 x 2304) * im2col(Xh[b]) (2304 x 3136)
// Block tile 128(M) x 128(N), K-step 32, 8 waves in a 4x2 grid,
// each wave: 2x4 grid of 16x16 tiles via v_wmma_f32_16x16x32_f16.
#define TM 128
#define TN 128
#define TK 32

__global__ __launch_bounds__(256) void conv_wmma_kernel(const _Float16* __restrict__ Xh,
                                                        const _Float16* __restrict__ Wh,
                                                        const float* __restrict__ aggB,
                                                        float* __restrict__ out) {
  __shared__ _Float16 sA[TM * TK];   // [m][k] row-major (TDM destination)
  __shared__ _Float16 sB[TN * TK];   // [n][k] (transposed im2col slab)

  const int b     = blockIdx.z;
  const int mBase = blockIdx.y * TM;
  const int nBase = blockIdx.x * TN;
  const int tid   = threadIdx.x;
  const int wave  = tid >> 5;
  const int lane  = tid & 31;
  const int wm    = wave >> 1;           // 0..3 -> M offset wm*32
  const int wn    = wave & 1;            // 0..1 -> N offset wn*64
  const int laneN = lane & 15;
  const int kof   = (lane < 16) ? 0 : 8; // ISA f16 A/B K-group selector

  const _Float16* Xb = Xh + (size_t)b * CIN * SP;
  const _Float16* Wb = Wh + (size_t)b * NTAP * COUT * CIN;

  // Precompute this thread's 16 B-staging slots (loop-invariant over tap/kb):
  // slot j -> LDS element (n,k); global offset base[j] = k*SP + y*HP + x.
  int bbase[16];
  int bok[16];
  for (int j = 0; j < 16; ++j) {
    const int e = tid + 256 * j;       // 0..4095
    const int n = e & (TN - 1);
    const int k = e >> 7;              // 0..31
    const int p = nBase + n;
    bok[j] = (p < PP);
    if (p < PP) {
      const int y = p / WW, x = p - y * WW;
      bbase[j] = k * SP + y * HP + x;
    } else {
      bbase[j] = 0;
    }
  }

  v8f acc[2][4] = {};

  for (int tap = 0; tap < NTAP; ++tap) {
    const int ky = tap / 3, kx = tap % 3;
    const int tapOfs = ky * HP + kx;
    const _Float16* Wt = Wb + (size_t)tap * COUT * CIN;

    for (int kb = 0; kb < CIN; kb += TK) {
      __syncthreads();   // previous iteration's fragment reads complete

      // --- stage A via Tensor Data Mover: 2D tile, 128 rows x 32 halves,
      //     row stride 256 halves, data_size = 2 bytes (D# per ISA ch.8) ---
      if (wave == 0) {
        const unsigned long long ga =
            (unsigned long long)(size_t)(Wt + (size_t)mBase * CIN + kb);
        const unsigned ldsa = (unsigned)(size_t)(&sA[0]);  // LDS byte offset
        v4u g0;
        g0[0] = 1u;                                  // count=1, user mode
        g0[1] = ldsa;                                // lds_addr
        g0[2] = (unsigned)(ga & 0xffffffffu);        // global_addr[31:0]
        g0[3] = (unsigned)(((ga >> 32) & 0x1ffffffu) // global_addr[56:32]
                           | 0x80000000u);           // type=2 ("image")
        v8i g1;
        g1[0] = 0x00010000;          // wg_mask=0, data_size=1 (2B)
        g1[1] = (int)(256u << 16);   // tensor_dim0 = 256 (lo16 in [31:16])
        g1[2] = (int)(256u << 16);   // dim0 hi=0 | tensor_dim1 = 256
        g1[3] = (int)(32u << 16);    // dim1 hi=0 | tile_dim0 = 32
        g1[4] = 128;                 // tile_dim1 = 128, tile_dim2 = 0
        g1[5] = 256;                 // tensor_dim0_stride = 256 (lo32)
        g1[6] = 0;                   // stride hi | tensor_dim1_stride lo
        g1[7] = 0;
        v4i g2; g2[0] = 0; g2[1] = 0; g2[2] = 0; g2[3] = 0;
        v4i g3; g3[0] = 0; g3[1] = 0; g3[2] = 0; g3[3] = 0;
#if __clang_major__ >= 23
        v8i g4; g4[0]=0; g4[1]=0; g4[2]=0; g4[3]=0; g4[4]=0; g4[5]=0; g4[6]=0; g4[7]=0;
        __builtin_amdgcn_tensor_load_to_lds(g0, g1, g2, g3, g4, 0);
#else
        __builtin_amdgcn_tensor_load_to_lds(g0, g1, g2, g3, 0);
#endif
      }

      // prefetch next K-slab of weights into cache
      if (kb + TK < CIN)
        __builtin_prefetch(Wt + (size_t)(mBase + (tid >> 1)) * CIN + kb + TK, 0, 1);

      // --- stage B transposed: sB[n][k] = Xh[cin=kb+k][y+ky][x+kx] ---
      const size_t kbOfs = (size_t)kb * SP + tapOfs;
      for (int j = 0; j < 16; ++j) {
        _Float16 v = (_Float16)0.f;
        if (bok[j]) v = Xb[kbOfs + bbase[j]];
        const int e = tid + 256 * j;
        sB[(e & (TN - 1)) * TK + (e >> 7)] = v;
      }

      if (wave == 0) __builtin_amdgcn_s_wait_tensorcnt(0);
      __syncthreads();

      // --- build fragments (two ds_load_b128 each, ISA K-group layout) ---
      v16h afrag[2], bfrag[4];
      for (int i = 0; i < 2; ++i) {
        const int row = wm * 32 + i * 16 + laneN;
        v8h lo = *(const v8h*)&sA[row * TK + kof];
        v8h hi = *(const v8h*)&sA[row * TK + 16 + kof];
        afrag[i] = __builtin_shufflevector(lo, hi, 0,1,2,3,4,5,6,7,8,9,10,11,12,13,14,15);
      }
      for (int j = 0; j < 4; ++j) {
        const int col = wn * 64 + j * 16 + laneN;
        v8h lo = *(const v8h*)&sB[col * TK + kof];
        v8h hi = *(const v8h*)&sB[col * TK + 16 + kof];
        bfrag[j] = __builtin_shufflevector(lo, hi, 0,1,2,3,4,5,6,7,8,9,10,11,12,13,14,15);
      }

      // --- 8 WMMAs, f32 accumulate ---
#pragma unroll
      for (int i = 0; i < 2; ++i)
#pragma unroll
        for (int j = 0; j < 4; ++j)
          acc[i][j] = __builtin_amdgcn_wmma_f32_16x16x32_f16(
              false, afrag[i], false, bfrag[j],
              (short)0, acc[i][j], false, false);
    }
  }

  // --- store D + bias; C/D layout: lane holds N=laneN, VGPR r -> M = r (+8 hi lanes)
  const int mo = (lane < 16) ? 0 : 8;
  for (int i = 0; i < 2; ++i) {
    const int m0 = mBase + wm * 32 + i * 16;
    for (int j = 0; j < 4; ++j) {
      const int n = nBase + wn * 64 + j * 16 + laneN;
      if (n < PP) {
        for (int r = 0; r < 8; ++r) {
          const int m = m0 + r + mo;
          out[((size_t)b * COUT + m) * PP + n] = acc[i][j][r] + aggB[b * COUT + m];
        }
      }
    }
  }
}

// ---------------------------------------------------------------------------
extern "C" void kernel_launch(void* const* d_in, const int* in_sizes, int n_in,
                              void* d_out, int out_size, void* d_ws, size_t ws_size,
                              hipStream_t stream) {
  const float* ref    = (const float*)d_in[0];
  const float* x      = (const float*)d_in[1];
  const float* fc1_w  = (const float*)d_in[2];
  const float* fc2_w  = (const float*)d_in[3];
  const float* fc2_b  = (const float*)d_in[4];
  const float* weight = (const float*)d_in[5];
  const float* bias   = (const float*)d_in[6];
  float* out = (float*)d_out;

  char* ws = (char*)d_ws;
  float*    pooled = (float*)(ws + WS_POOLED);
  float*    attn   = (float*)(ws + WS_ATTN);
  float*    aggB   = (float*)(ws + WS_AGGB);
  _Float16* Xh     = (_Float16*)(ws + WS_XH);
  _Float16* Wh     = (_Float16*)(ws + WS_WH);

  // 1) pooling: one block per (b, c)
  pool_kernel<<<NB * CREF, 256, 0, stream>>>(ref, pooled);

  // 2) attention + aggregated bias
  attn_kernel<<<1, 256, 0, stream>>>(pooled, fc1_w, fc2_w, fc2_b, bias, attn, aggB);

  // 3) padded f16 activations
  {
    const size_t total = (size_t)NB * CIN * SP;
    cvt_x_kernel<<<(unsigned)((total + 255) / 256), 256, 0, stream>>>(x, Xh);
  }

  // 4) aggregated tap-major f16 weights
  {
    const size_t total = (size_t)NB * NTAP * COUT * CIN;
    agg_w_kernel<<<(unsigned)((total + 255) / 256), 256, 0, stream>>>(weight, attn, Wh);
  }

  // 5) WMMA implicit-GEMM conv: grid (ceil(3136/128)=25, 256/128=2, B=16)
  dim3 grid((PP + TN - 1) / TN, COUT / TM, NB);
  conv_wmma_kernel<<<grid, 256, 0, stream>>>(Xh, Wh, aggB, out);
}
